// MyViT_70695161692421
// MI455X (gfx1250) — matrix operand order repaired and checked by reading, not compiled, not run on hardware
//
#include <hip/hip_runtime.h>
#include <hip/hip_bf16.h>

// ---------------- problem constants ----------------
#define B_      64
#define SEQ_    197
#define SP_     224      // SEQ padded to multiple of 32 (7*32)
#define D_      768
#define NH_     12
#define DH_     64
#define IND_    768      // C * 16 * 16
#define NPATCH_ 196
#define OUTD_   1000
#define OUTP_   1008     // padded to multiple of 16

typedef __attribute__((ext_vector_type(16))) _Float16 v16h;
typedef __attribute__((ext_vector_type(8)))  _Float16 v8h;
typedef __attribute__((ext_vector_type(8)))  float    v8f;

// WMMA: D = A(16x32 f16) * B(32x16 f16) + C(16x16 f32)
static __device__ inline v8f wmma32(v16h a, v16h b, v8f c) {
  return __builtin_amdgcn_wmma_f32_16x16x32_f16(false, a, false, b, (short)0, c,
                                                false, false);
}

// A/B fragment from f16 memory: two chunks (16B + 16B)
static __device__ inline v16h frag_f16(const _Float16* p0, const _Float16* p1) {
  v16h r;
  *(v8h*)&r       = *(const v8h*)p0;
  *((v8h*)&r + 1) = *(const v8h*)p1;
  return r;
}

// ---------------- positional-embedding table (197 x 768), computed once ---
__global__ void k_posemb(float* __restrict__ pe) {
  int idx = blockIdx.x * 256 + threadIdx.x;            // SEQ_*D_ (divisible)
  int i = idx / D_, j = idx % D_;
  // exponent = (j - j%2)/768 ; angle = i * 10000^-exponent
  float expo  = (float)(j & ~1) * (1.0f / (float)D_);
  // 10000^-e = exp2(-e * log2(10000))
  float angle = (float)i * exp2f(-expo * 13.287712379549449f);
  pe[idx] = (j & 1) ? __cosf(angle) : __sinf(angle);
}

// ---------------- weight prep: fp32 (K,N) -> f16 (Np,K), zero-pad rows ----
__global__ void k_transpose(const float* __restrict__ in, _Float16* __restrict__ out,
                            int K, int N, int Np) {
  long idx = (long)blockIdx.x * 256 + threadIdx.x;
  if (idx >= (long)Np * K) return;
  int n = (int)(idx / K), k = (int)(idx % K);
  out[idx] = (n < N) ? (_Float16)in[(size_t)k * N + n] : (_Float16)0.f;
}

// per-head (NH,DH,DH) transpose of wq/wk/wv, fp32 -> f16
__global__ void k_theads(const float* __restrict__ wq, const float* __restrict__ wk,
                         const float* __restrict__ wv,
                         _Float16* oq, _Float16* ok, _Float16* ov) {
  int idx = blockIdx.x * 256 + threadIdx.x;            // 3*12*64*64
  int which = idx / (NH_ * DH_ * DH_);
  int rem   = idx % (NH_ * DH_ * DH_);
  int h = rem / (DH_ * DH_);
  int r2 = rem % (DH_ * DH_);
  int e = r2 / DH_, d = r2 % DH_;
  const float* in = (which == 0) ? wq : (which == 1) ? wk : wv;
  _Float16*    out = (which == 0) ? oq : (which == 1) ? ok : ov;
  out[rem] = (_Float16)in[(size_t)h * DH_ * DH_ + (size_t)d * DH_ + e];
}

// ---------------- patch embedding GEMM: (12544 x 768) @ (768 x 768) -------
// one wave -> 16-row x 64-col strip; A converted f32->f16 on the fly
__global__ void __launch_bounds__(128)
k_embed(const float* __restrict__ img, const _Float16* __restrict__ wT,
        const float* __restrict__ bmap, const float* __restrict__ pe,
        float* __restrict__ tokens) {
  int wid  = blockIdx.x * 4 + (threadIdx.x >> 5);      // 784*12 waves
  int lane = threadIdx.x & 31;
  int mt = wid / 12, strip = wid % 12;
  int ml = lane & 15, hi = lane >> 4;
  int ka = hi ? 8 : 0, kb = hi ? 16 : 0, mb = hi ? 8 : 0;

  const float* arow = img + (size_t)(mt * 16 + ml) * IND_;
  const _Float16* bt[4];
#pragma unroll
  for (int nt = 0; nt < 4; ++nt)
    bt[nt] = wT + (size_t)(strip * 64 + nt * 16 + ml) * IND_;

  v8f acc[4] = {};
  for (int k = 0; k < IND_; k += 32) {
    // stream hint: next k-step of the image row (images are the only
    // operand too large for L2 reuse)
    __builtin_prefetch(arow + k + 128, 0, 1);
    v16h a;
#pragma unroll
    for (int i = 0; i < 8; ++i) {
      a[i]     = (_Float16)arow[k + ka + i];
      a[8 + i] = (_Float16)arow[k + ka + 16 + i];
    }
#pragma unroll
    for (int nt = 0; nt < 4; ++nt) {
      v16h b = frag_f16(bt[nt] + k + ka, bt[nt] + k + ka + 16);
      acc[nt] = wmma32(a, b, acc[nt]);
    }
  }
#pragma unroll
  for (int nt = 0; nt < 4; ++nt) {
#pragma unroll
    for (int r = 0; r < 8; ++r) {
      int grow = mt * 16 + mb + r;                     // flat patch row
      int b = grow / NPATCH_, p = grow % NPATCH_;
      int i = p + 1;
      int n = strip * 64 + nt * 16 + ml;
      tokens[((size_t)b * SEQ_ + i) * D_ + n] =
          acc[nt][r] + bmap[n] + pe[i * D_ + n];
    }
  }
}

// cls token row + its positional embedding row
__global__ void k_clspos(const float* __restrict__ cls, const float* __restrict__ pe,
                         float* __restrict__ tokens) {
  int idx = blockIdx.x * 256 + threadIdx.x;            // 64*768
  int b = idx / D_, n = idx % D_;
  tokens[(size_t)b * SEQ_ * D_ + n] = cls[n] + pe[n];
}

// ---------------- 2-D layernorm (over whole SEQ*D per batch) --------------
__global__ void k_lnstats(const float* __restrict__ tokens, float* __restrict__ stats) {
  __shared__ float rs[256], rq[256];
  int b = blockIdx.x, tid = threadIdx.x;
  const float* p = tokens + (size_t)b * SEQ_ * D_;
  float s = 0.f, q = 0.f;
  for (int i = tid; i < SEQ_ * D_; i += 256) { float x = p[i]; s += x; q += x * x; }
  rs[tid] = s; rq[tid] = q; __syncthreads();
  for (int st = 128; st > 0; st >>= 1) {
    if (tid < st) { rs[tid] += rs[tid + st]; rq[tid] += rq[tid + st]; }
    __syncthreads();
  }
  if (tid == 0) {
    float inv = 1.0f / (float)(SEQ_ * D_);
    float mu = rs[0] * inv;
    float var = rq[0] * inv - mu * mu;
    stats[2 * b] = mu; stats[2 * b + 1] = rsqrtf(var + 1e-5f);
  }
}

__global__ void k_lnapply(const float* __restrict__ tokens, const float* __restrict__ g,
                          const float* __restrict__ bb, const float* __restrict__ stats,
                          _Float16* __restrict__ out) {
  size_t idx = (size_t)blockIdx.x * 256 + threadIdx.x; // B * 224 * 768
  int b = (int)(idx / ((size_t)SP_ * D_));
  int rem = (int)(idx % ((size_t)SP_ * D_));
  int s = rem / D_, d = rem % D_;
  float v = 0.f;
  if (s < SEQ_) {
    float mu = stats[2 * b], rstd = stats[2 * b + 1];
    v = (tokens[((size_t)b * SEQ_ + s) * D_ + d] - mu) * rstd * g[s * D_ + d] +
        bb[s * D_ + d];
  }
  out[idx] = (_Float16)v;
}

// ---------------- QKV projections: per-head 64x64 GEMMs -------------------
// q,k: (B,NH,SP,DH) f16 ; v stored transposed: (B,NH,DH,SP) f16
__global__ void __launch_bounds__(128)
k_qkv(const _Float16* __restrict__ x,
      const _Float16* __restrict__ wqT, const _Float16* __restrict__ wkT,
      const _Float16* __restrict__ wvT,
      const float* __restrict__ bq, const float* __restrict__ bk,
      const float* __restrict__ bv,
      _Float16* __restrict__ qh, _Float16* __restrict__ kh,
      _Float16* __restrict__ vT) {
  int wid  = blockIdx.x * 4 + (threadIdx.x >> 5);      // 64*14*12 waves
  int lane = threadIdx.x & 31;
  int b = wid / (14 * NH_);
  int rem = wid % (14 * NH_);
  int mt = rem / NH_, h = rem % NH_;
  int ml = lane & 15, hi = lane >> 4;
  int ka = hi ? 8 : 0, kb = hi ? 16 : 0, mb = hi ? 8 : 0;

  const _Float16* arow =
      x + ((size_t)b * SP_ + mt * 16 + ml) * D_ + h * DH_;
  v16h a0 = frag_f16(arow + ka,      arow + ka + 16);
  v16h a1 = frag_f16(arow + 32 + ka, arow + 32 + ka + 16);

  for (int which = 0; which < 3; ++which) {
    const _Float16* wT = (which == 0) ? wqT : (which == 1) ? wkT : wvT;
    const float* bias  = (which == 0) ? bq  : (which == 1) ? bk  : bv;
#pragma unroll
    for (int nt = 0; nt < 4; ++nt) {
      int n = nt * 16 + ml;
      const _Float16* bt = wT + ((size_t)h * DH_ + n) * DH_;
      v16h b0 = frag_f16(bt + ka,      bt + ka + 16);
      v16h b1 = frag_f16(bt + 32 + ka, bt + 32 + ka + 16);
      v8f acc = {};
      acc = wmma32(a0, b0, acc);
      acc = wmma32(a1, b1, acc);
      float bs = bias[h * DH_ + n];
      if (which < 2) {
        _Float16* dst = ((which == 0) ? qh : kh) +
                        (((size_t)(b * NH_ + h)) * SP_ + mt * 16) * DH_;
#pragma unroll
        for (int r = 0; r < 8; ++r)
          dst[(size_t)(mb + r) * DH_ + n] = (_Float16)(acc[r] + bs);
      } else {
        _Float16* dst = vT + (((size_t)(b * NH_ + h)) * DH_ + n) * SP_ + mt * 16;
#pragma unroll
        for (int r = 0; r < 8; ++r)
          dst[mb + r] = (_Float16)(acc[r] + bs);
      }
    }
  }
}

// ---------------- fused attention per (b,h): scores->softmax->AV in LDS ---
__global__ void __launch_bounds__(256)
k_attn(const _Float16* __restrict__ qh, const _Float16* __restrict__ kh,
       const _Float16* __restrict__ vT, float* __restrict__ tokens) {
  extern __shared__ float s[];                         // 224*224 f32 = 196 KB
  int b = blockIdx.x / NH_, h = blockIdx.x % NH_;
  int wvi  = threadIdx.x >> 5;
  int lane = threadIdx.x & 31;
  int ml = lane & 15, hi = lane >> 4;
  int ka = hi ? 8 : 0, kb = hi ? 16 : 0, mb = hi ? 8 : 0;

  const _Float16* qb   = qh + (size_t)(b * NH_ + h) * SP_ * DH_;
  const _Float16* kbse = kh + (size_t)(b * NH_ + h) * SP_ * DH_;
  const _Float16* vb   = vT + (size_t)(b * NH_ + h) * DH_ * SP_;

  // phase 1: S = Q K^T * 1/8, 14x14 tiles
  for (int t = wvi; t < 14 * 14; t += 8) {
    int mi = t / 14, ti = t % 14;
    const _Float16* ar = qb + (size_t)(mi * 16 + ml) * DH_;
    v16h a0 = frag_f16(ar + ka,      ar + ka + 16);
    v16h a1 = frag_f16(ar + 32 + ka, ar + 32 + ka + 16);
    const _Float16* br = kbse + (size_t)(ti * 16 + ml) * DH_;
    v16h b0 = frag_f16(br + ka,      br + ka + 16);
    v16h b1 = frag_f16(br + 32 + ka, br + 32 + ka + 16);
    v8f acc = {};
    acc = wmma32(a0, b0, acc);
    acc = wmma32(a1, b1, acc);
#pragma unroll
    for (int r = 0; r < 8; ++r)
      s[(mi * 16 + mb + r) * SP_ + ti * 16 + ml] = acc[r] * 0.125f;
  }
  __syncthreads();

  // phase 2: row softmax over valid t<SEQ, zero the padding
  int row = threadIdx.x;
  if (row < SP_) {
    float* pr = s + row * SP_;
    if (row < SEQ_) {
      float mx = -1e30f;
      for (int j = 0; j < SEQ_; ++j) mx = fmaxf(mx, pr[j]);
      float sum = 0.f;
      for (int j = 0; j < SEQ_; ++j) { float e = __expf(pr[j] - mx); pr[j] = e; sum += e; }
      float inv = 1.0f / sum;
      for (int j = 0; j < SEQ_; ++j) pr[j] *= inv;
      for (int j = SEQ_; j < SP_; ++j) pr[j] = 0.f;
    } else {
      for (int j = 0; j < SP_; ++j) pr[j] = 0.f;
    }
  }
  __syncthreads();

  // phase 3: O = A V, accumulate into tokens (residual add)
  for (int t = wvi; t < 14 * 4; t += 8) {
    int mi = t >> 2, nt = t & 3;
    const float* arow = s + (mi * 16 + ml) * SP_;
    const _Float16* br = vb + (size_t)(nt * 16 + ml) * SP_;
    v8f acc = {};
    for (int kt = 0; kt < SP_; kt += 32) {
      v16h a;
      const float* p0 = arow + kt + ka;
      const float* p1 = arow + kt + ka + 16;
#pragma unroll
      for (int i = 0; i < 8; ++i) { a[i] = (_Float16)p0[i]; a[8 + i] = (_Float16)p1[i]; }
      v16h bf = frag_f16(br + kt + ka, br + kt + ka + 16);
      acc = wmma32(a, bf, acc);
    }
#pragma unroll
    for (int r = 0; r < 8; ++r) {
      int sr = mi * 16 + mb + r;
      if (sr < SEQ_) {
        int c = h * DH_ + nt * 16 + ml;
        tokens[((size_t)b * SEQ_ + sr) * D_ + c] += acc[r];
      }
    }
  }
}

// ---------------- MLP: out += relu(y @ w_enc + b_enc) ---------------------
__global__ void __launch_bounds__(128)
k_mlp(const _Float16* __restrict__ y, const _Float16* __restrict__ wT,
      const float* __restrict__ bias, float* __restrict__ tokens) {
  int wid  = blockIdx.x * 4 + (threadIdx.x >> 5);      // 896*12 waves
  int lane = threadIdx.x & 31;
  int mt = wid / 12, strip = wid % 12;
  int ml = lane & 15, hi = lane >> 4;
  int ka = hi ? 8 : 0, kb = hi ? 16 : 0, mb = hi ? 8 : 0;

  const _Float16* arow = y + (size_t)(mt * 16 + ml) * D_;
  const _Float16* bt[4];
#pragma unroll
  for (int nt = 0; nt < 4; ++nt)
    bt[nt] = wT + (size_t)(strip * 64 + nt * 16 + ml) * D_;

  v8f acc[4] = {};
  for (int k = 0; k < D_; k += 32) {
    __builtin_prefetch(arow + k + 64, 0, 1);
    v16h a = frag_f16(arow + k + ka, arow + k + ka + 16);
#pragma unroll
    for (int nt = 0; nt < 4; ++nt) {
      v16h b = frag_f16(bt[nt] + k + ka, bt[nt] + k + ka + 16);
      acc[nt] = wmma32(a, b, acc[nt]);
    }
  }
#pragma unroll
  for (int nt = 0; nt < 4; ++nt) {
#pragma unroll
    for (int r = 0; r < 8; ++r) {
      int grow = mt * 16 + mb + r;
      int b = grow / SP_, sr = grow % SP_;
      if (sr < SEQ_) {
        int c = strip * 64 + nt * 16 + ml;
        tokens[((size_t)b * SEQ_ + sr) * D_ + c] += fmaxf(acc[nt][r] + bias[c], 0.f);
      }
    }
  }
}

// ---------------- classifier head ----------------------------------------
__global__ void k_clsext(const float* __restrict__ tokens, _Float16* __restrict__ cls) {
  int idx = blockIdx.x * 256 + threadIdx.x;            // 64*768
  int b = idx / D_, k = idx % D_;
  cls[idx] = (_Float16)tokens[(size_t)b * SEQ_ * D_ + k];
}

__global__ void __launch_bounds__(128)
k_head(const _Float16* __restrict__ cls, const _Float16* __restrict__ wT,
       const float* __restrict__ bias, float* __restrict__ logits) {
  int wid  = blockIdx.x * 4 + (threadIdx.x >> 5);      // 4*63 waves
  int lane = threadIdx.x & 31;
  int mt = wid / 63, nt = wid % 63;
  int ml = lane & 15, hi = lane >> 4;
  int ka = hi ? 8 : 0, kb = hi ? 16 : 0, mb = hi ? 8 : 0;

  const _Float16* arow = cls + (size_t)(mt * 16 + ml) * D_;
  const _Float16* bt   = wT + (size_t)(nt * 16 + ml) * D_;
  v8f acc = {};
  for (int k = 0; k < D_; k += 32) {
    v16h a = frag_f16(arow + k + ka, arow + k + ka + 16);
    v16h b = frag_f16(bt + k + ka, bt + k + ka + 16);
    acc = wmma32(a, b, acc);
  }
  int n = nt * 16 + ml;
  if (n < OUTD_) {
#pragma unroll
    for (int r = 0; r < 8; ++r)
      logits[(size_t)(mt * 16 + mb + r) * OUTD_ + n] = acc[r] + bias[n];
  }
}

__global__ void k_softmax(const float* __restrict__ logits, float* __restrict__ out) {
  __shared__ float red[256];
  int b = blockIdx.x, tid = threadIdx.x;
  const float* lr = logits + (size_t)b * OUTD_;
  float mx = -1e30f;
  for (int j = tid; j < OUTD_; j += 256) mx = fmaxf(mx, lr[j]);
  red[tid] = mx; __syncthreads();
  for (int st = 128; st > 0; st >>= 1) {
    if (tid < st) red[tid] = fmaxf(red[tid], red[tid + st]);
    __syncthreads();
  }
  mx = red[0]; __syncthreads();
  float sum = 0.f;
  for (int j = tid; j < OUTD_; j += 256) sum += __expf(lr[j] - mx);
  red[tid] = sum; __syncthreads();
  for (int st = 128; st > 0; st >>= 1) {
    if (tid < st) red[tid] += red[tid + st];
    __syncthreads();
  }
  float inv = 1.0f / red[0];
  for (int j = tid; j < OUTD_; j += 256)
    out[(size_t)b * OUTD_ + j] = __expf(lr[j] - mx) * inv;
}

// ---------------- host launcher -------------------------------------------
extern "C" void kernel_launch(void* const* d_in, const int* in_sizes, int n_in,
                              void* d_out, int out_size, void* d_ws, size_t ws_size,
                              hipStream_t stream) {
  const float* images  = (const float*)d_in[0];
  const float* w_map   = (const float*)d_in[1];
  const float* b_map   = (const float*)d_in[2];
  const float* cls_tok = (const float*)d_in[3];
  const float* ln1_g   = (const float*)d_in[4];
  const float* ln1_b   = (const float*)d_in[5];
  const float* wq      = (const float*)d_in[6];
  const float* bq      = (const float*)d_in[7];
  const float* wk      = (const float*)d_in[8];
  const float* bk      = (const float*)d_in[9];
  const float* wv      = (const float*)d_in[10];
  const float* bv      = (const float*)d_in[11];
  const float* ln2_g   = (const float*)d_in[12];
  const float* ln2_b   = (const float*)d_in[13];
  const float* w_enc   = (const float*)d_in[14];
  const float* b_enc   = (const float*)d_in[15];
  const float* w_head  = (const float*)d_in[16];
  const float* b_head  = (const float*)d_in[17];

  uintptr_t base = (uintptr_t)d_ws;
  size_t off = 0;
  auto alloc = [&](size_t bytes) -> void* {
    off = (off + 255) & ~(size_t)255;
    void* p = (void*)(base + off);
    off += bytes;
    return p;
  };

  float*     tokens = (float*)    alloc((size_t)B_ * SEQ_ * D_ * 4);
  float*     pe     = (float*)    alloc((size_t)SEQ_ * D_ * 4);
  _Float16*  wmapT  = (_Float16*) alloc((size_t)D_ * IND_ * 2);
  _Float16*  wencT  = (_Float16*) alloc((size_t)D_ * D_ * 2);
  _Float16*  wheadT = (_Float16*) alloc((size_t)OUTP_ * D_ * 2);
  _Float16*  wqT    = (_Float16*) alloc((size_t)NH_ * DH_ * DH_ * 2);
  _Float16*  wkT    = (_Float16*) alloc((size_t)NH_ * DH_ * DH_ * 2);
  _Float16*  wvT    = (_Float16*) alloc((size_t)NH_ * DH_ * DH_ * 2);
  _Float16*  x16    = (_Float16*) alloc((size_t)B_ * SP_ * D_ * 2);
  _Float16*  q16    = (_Float16*) alloc((size_t)B_ * NH_ * SP_ * DH_ * 2);
  _Float16*  k16    = (_Float16*) alloc((size_t)B_ * NH_ * SP_ * DH_ * 2);
  _Float16*  vT16   = (_Float16*) alloc((size_t)B_ * NH_ * DH_ * SP_ * 2);
  float*     stats  = (float*)    alloc((size_t)B_ * 2 * 4);
  _Float16*  cls16  = (_Float16*) alloc((size_t)B_ * D_ * 2);
  float*     logits = (float*)    alloc((size_t)B_ * OUTD_ * 4);

  // one-time tables / weight prep (f16, transposed)
  k_posemb<<<(SEQ_ * D_) / 256, 256, 0, stream>>>(pe);
  k_transpose<<<(D_ * IND_ + 255) / 256, 256, 0, stream>>>(w_map, wmapT, IND_, D_, D_);
  k_transpose<<<(D_ * D_ + 255) / 256, 256, 0, stream>>>(w_enc, wencT, D_, D_, D_);
  k_transpose<<<(OUTP_ * D_ + 255) / 256, 256, 0, stream>>>(w_head, wheadT, D_, OUTD_, OUTP_);
  k_theads<<<(3 * NH_ * DH_ * DH_) / 256, 256, 0, stream>>>(wq, wk, wv, wqT, wkT, wvT);

  // patch embed + cls + posemb
  k_embed<<<(784 * 12) / 4, 128, 0, stream>>>(images, wmapT, b_map, pe, tokens);
  k_clspos<<<(B_ * D_) / 256, 256, 0, stream>>>(cls_tok, pe, tokens);

  // LN1 -> x16
  k_lnstats<<<B_, 256, 0, stream>>>(tokens, stats);
  k_lnapply<<<(B_ * SP_ * D_) / 256, 256, 0, stream>>>(tokens, ln1_g, ln1_b, stats, x16);

  // QKV
  k_qkv<<<(B_ * 14 * NH_) / 4, 128, 0, stream>>>(x16, wqT, wkT, wvT, bq, bk, bv,
                                                 q16, k16, vT16);

  // fused attention, residual into tokens; 196 KB LDS per workgroup
  (void)hipFuncSetAttribute((const void*)k_attn,
                            hipFuncAttributeMaxDynamicSharedMemorySize,
                            SP_ * SP_ * 4);
  k_attn<<<B_ * NH_, 256, SP_ * SP_ * 4, stream>>>(q16, k16, vT16, tokens);

  // LN2 -> reuse x16 as y
  k_lnstats<<<B_, 256, 0, stream>>>(tokens, stats);
  k_lnapply<<<(B_ * SP_ * D_) / 256, 256, 0, stream>>>(tokens, ln2_g, ln2_b, stats, x16);

  // MLP, residual into tokens
  k_mlp<<<(B_ * 14 * 12) / 4, 128, 0, stream>>>(x16, wencT, b_enc, tokens);

  // head + softmax
  k_clsext<<<(B_ * D_) / 256, 256, 0, stream>>>(tokens, cls16);
  k_head<<<(4 * 63 + 3) / 4, 128, 0, stream>>>(cls16, wheadT, b_head, logits);
  k_softmax<<<B_, 256, 0, stream>>>(logits, (float*)d_out);

  (void)in_sizes; (void)n_in; (void)out_size; (void)ws_size;
}
